// Net_21947282882969
// MI455X (gfx1250) — matrix operand order repaired and checked by reading, not compiled, not run on hardware
//
#include <hip/hip_runtime.h>

#define LEN 5
#define DIM 200
#define INV_TEMP 100.0f
#define N_SINKHORN 20

typedef __attribute__((ext_vector_type(16))) _Float16 v16h;
typedef __attribute__((ext_vector_type(8)))  float    v8f;

// Load 8 consecutive floats p[base..base+7] as two float4s.
// If guard (tail K-chunk), clamp addresses into bounds; values past DIM are
// garbage-but-finite, which is harmless (the B-side is zeroed there).
__device__ __forceinline__ void load8(const float* __restrict__ p, int base,
                                      bool guard, float v[8]) {
    #pragma unroll
    for (int q = 0; q < 2; ++q) {
        int idx = base + q * 4;
        if (guard && idx > DIM - 4) idx = DIM - 4;
        const float4 t = *(const float4*)(p + idx);
        v[q * 4 + 0] = t.x; v[q * 4 + 1] = t.y;
        v[q * 4 + 2] = t.z; v[q * 4 + 3] = t.w;
    }
}

// One wave32 does everything:
//   trans = (x*Wc + bc) @ W2^T + b2  via 7x V_WMMA_F32_16X16X32_F16 (K 200->224)
//   20 Sinkhorn iterations on the 5x5 held in registers
//   out = exp(s) @ x
__global__ __launch_bounds__(32) void sinkhorn5_kernel(
    const float* __restrict__ x,    // [5]
    const float* __restrict__ Wc,   // [200,1] flat
    const float* __restrict__ bc,   // [200]
    const float* __restrict__ W2,   // [5,200] row-major
    const float* __restrict__ b2,   // [5]
    float* __restrict__ out)        // [5]
{
    const int lane = threadIdx.x & 31;
    const int half = lane >> 4;           // 16-lane half
    const int m    = lane & 15;           // A-row (M) / B-col (N)
    const int mr   = (m < LEN) ? m : 0;   // clamped: rows/cols >=5 carry garbage, never read
    const float xm = x[mr];
    const float* __restrict__ w2row = W2 + mr * DIM;

    // ---- GEMM: C[16x16] += A[16x32] * B[32x16], 7 K-chunks of 32 ----
    // A (16-bit 16x32): lane holds row M=m; elem e -> K = kb + half*16 + e
    // B (16-bit 32x16): lane holds column N=m; elem e -> K = kb + half*16 + e
    v8f c = {};
    #pragma unroll
    for (int kb = 0; kb < 224; kb += 32) {
        const bool tail = (kb == 192);        // compile-time per unrolled iteration
        const int a0 = kb + half * 16;        // A run 0 (8 K-values)
        const int a1 = kb + half * 16 + 8;    // A run 1
        const int b0 = kb + half * 16;        // B run (16 K-values)

        float wc0[8], wc1[8], bc0[8], bc1[8], w2v[16];
        load8(Wc, a0, tail, wc0);
        load8(Wc, a1, tail, wc1);
        load8(bc, a0, tail, bc0);
        load8(bc, a1, tail, bc1);
        load8(w2row, b0,     tail, w2v);
        load8(w2row, b0 + 8, tail, w2v + 8);

        v16h a, b;
        #pragma unroll
        for (int e = 0; e < 8; ++e) {
            a[e]     = (_Float16)fmaf(xm, wc0[e], bc0[e]);   // embs_x on the fly
            a[e + 8] = (_Float16)fmaf(xm, wc1[e], bc1[e]);
        }
        #pragma unroll
        for (int e = 0; e < 16; ++e) {
            float bv = w2v[e];
            if (tail) bv = (b0 + e < DIM) ? bv : 0.0f;  // zero B side of K-padding
            b[e] = (_Float16)bv;
        }
        // 8 args: (neg_a, A, neg_b, B, c_mod, C, reuse_a, reuse_b)
        c = __builtin_amdgcn_wmma_f32_16x16x32_f16(false, a, false, b, (short)0, c,
                                                   false, false);
    }

    // C layout: lane = column N, rows M=i in c[i] (lanes<16). s[i][j] lives in lane j.
    const float b2j = b2[mr];

    float s[LEN];
    #pragma unroll
    for (int i = 0; i < LEN; ++i) s[i] = (c[i] + b2j) * INV_TEMP;

    // ---- 20 serial Sinkhorn iterations ----
    for (int it = 0; it < N_SINKHORN; ++it) {
        // column logsumexp (axis=0): lane-local over registers
        float mc = s[0];
        #pragma unroll
        for (int i = 1; i < LEN; ++i) mc = fmaxf(mc, s[i]);
        float sc = 0.0f;
        #pragma unroll
        for (int i = 0; i < LEN; ++i) sc += __expf(s[i] - mc);
        const float lc = mc + __logf(sc);
        #pragma unroll
        for (int i = 0; i < LEN; ++i) s[i] -= lc;

        // row logsumexp (axis=1): across lanes 0..4
        #pragma unroll
        for (int i = 0; i < LEN; ++i) {
            const float t0 = __shfl(s[i], 0, 32);
            const float t1 = __shfl(s[i], 1, 32);
            const float t2 = __shfl(s[i], 2, 32);
            const float t3 = __shfl(s[i], 3, 32);
            const float t4 = __shfl(s[i], 4, 32);
            const float mx = fmaxf(fmaxf(fmaxf(t0, t1), fmaxf(t2, t3)), t4);
            const float sr = __expf(t0 - mx) + __expf(t1 - mx) + __expf(t2 - mx) +
                             __expf(t3 - mx) + __expf(t4 - mx);
            s[i] -= mx + __logf(sr);
        }
    }

    // ---- out[i] = sum_{j=0..4} exp(s[i][j]) * x[j]  (lane j holds column j) ----
    float p[LEN];
    #pragma unroll
    for (int i = 0; i < LEN; ++i) p[i] = __expf(s[i]) * xm;

    float r[LEN];
    #pragma unroll
    for (int i = 0; i < LEN; ++i)
        r[i] = __shfl(p[i], 0, 32) + __shfl(p[i], 1, 32) + __shfl(p[i], 2, 32) +
               __shfl(p[i], 3, 32) + __shfl(p[i], 4, 32);

    if (lane == 0) {
        #pragma unroll
        for (int i = 0; i < LEN; ++i) out[i] = r[i];
    }
}

extern "C" void kernel_launch(void* const* d_in, const int* in_sizes, int n_in,
                              void* d_out, int out_size, void* d_ws, size_t ws_size,
                              hipStream_t stream) {
    (void)in_sizes; (void)n_in; (void)out_size; (void)d_ws; (void)ws_size;
    const float* x  = (const float*)d_in[0];  // [5]
    const float* Wc = (const float*)d_in[1];  // [200,1]
    const float* bc = (const float*)d_in[2];  // [200]
    const float* W2 = (const float*)d_in[3];  // [5,200]
    const float* b2 = (const float*)d_in[4];  // [5]
    sinkhorn5_kernel<<<1, 32, 0, stream>>>(x, Wc, bc, W2, b2, (float*)d_out);
}
